// MultiHeadAttentionWithRelativePositionBias_36859409334607
// MI455X (gfx1250) — compile-verified, matrix operand order checked by reading, not run on hardware
//
#include <hip/hip_runtime.h>

// ---------------------------------------------------------------------------
// MHA with relative position bias, CDNA5 (gfx1250, wave32, WMMA 16x16x32 f16)
//
//   1. gemm_bias : LDS-tiled (64x128 block, 2x2 WMMA/wave, double-buffered)
//   2. logits    : QK^T with 1x4 k-tile register blocking + bias epilogue
//   3. softmax   : row-wise, in place over attn
//   4. attnv     : attn streamed once; V transposed into LDS, 4 WMMA/wave
//   5. gemm_bias : output dense
// ---------------------------------------------------------------------------

typedef __attribute__((ext_vector_type(16))) _Float16 v16h;
typedef __attribute__((ext_vector_type(8)))  _Float16 v8h;
typedef __attribute__((ext_vector_type(8)))  float    v8f;

#define D_MODEL   1024
#define NUM_HEADS 16
#define DEPTH     64
#define MAX_SEQ   2048
#define BATCH     2
#define SEQ       2048
#define BS        (BATCH * SEQ)   // 4096 rows for the dense GEMMs

#define BM  64                    // GEMM block tile M
#define BN  128                   // GEMM block tile N
#define BK  32                    // GEMM k-step (one WMMA K)
#define LDP 48                    // LDS row pitch in halves (96B: 16B-aligned,
                                  // 24-bank skew between consecutive rows)

#define WMMA_F16(A, B, C) \
  __builtin_amdgcn_wmma_f32_16x16x32_f16(false, (A), false, (B), (short)0, (C), false, false)

// ---- WMMA fragment helpers (ISA 7.12.2 layouts, wave32) --------------------
// 16-bit A 16x32: lanes 0-15 hold row M=lane with halves 0..7 = K 0..7 and
// 8..15 = K 16..23; lanes 16-31 hold the K 8..15 / 24..31 halves. B mirrors
// with lane = column.

// Fragment from row-major f32 in global memory (ld in floats).
__device__ __forceinline__ v16h frag_row(const float* __restrict__ src, int ld,
                                         int row, int k0, int lane) {
  const int half = (lane >> 4) << 3;                 // 0 or 8
  const float* p = src + (size_t)row * ld + k0 + half;
  v16h f;
#pragma unroll
  for (int j = 0; j < 8; ++j) f[j]     = (_Float16)p[j];
#pragma unroll
  for (int j = 0; j < 8; ++j) f[8 + j] = (_Float16)p[16 + j];
  return f;
}

// Fragment from an LDS tile row (pitch LDP halves); p = &tile[row][half].
// Two 16B ds_load_b128 (addresses are 16B aligned: row*96 + {0,16}).
__device__ __forceinline__ v16h frag_lds(const _Float16* p) {
  v8h lo = *(const v8h*)(p);
  v8h hi = *(const v8h*)(p + 16);
  v16h f;
#pragma unroll
  for (int j = 0; j < 8; ++j) { f[j] = lo[j]; f[8 + j] = hi[j]; }
  return f;
}

// ---- Y[M,N] = X[M,K] @ W[N,K]^T + bias  (LDS tiled, double buffered) -------
__global__ __launch_bounds__(256)
void gemm_bias_kernel(const float* __restrict__ X, const float* __restrict__ W,
                      const float* __restrict__ bias, float* __restrict__ Y,
                      int M, int N, int K) {
  __shared__ _Float16 As[2][BM][LDP];   // 12 KB
  __shared__ _Float16 Bs[2][BN][LDP];   // 24 KB
  const int t    = threadIdx.x;
  const int lane = t & 31;
  const int wave = t >> 5;                   // 0..7 -> 2x4 wave grid
  const int wm   = (wave >> 2) << 5;         // 0 / 32
  const int wn   = (wave & 3) << 5;          // 0 / 32 / 64 / 96
  const int nb   = N / BN;
  const int bm0  = (blockIdx.x / nb) * BM;
  const int bn0  = (blockIdx.x % nb) * BN;

  const int lr = t >> 2;                     // loader row   0..63
  const int lc = (t & 3) << 3;               // loader col0  0,8,16,24

  auto loadA = [&](int buf, int k0) {
    const float* p = X + (size_t)(bm0 + lr) * K + k0 + lc;
    v8h h;
#pragma unroll
    for (int j = 0; j < 8; ++j) h[j] = (_Float16)p[j];
    *(v8h*)&As[buf][lr][lc] = h;
  };
  auto loadB = [&](int buf, int k0) {
#pragma unroll
    for (int i = 0; i < 2; ++i) {
      const float* p = W + (size_t)(bn0 + (i << 6) + lr) * K + k0 + lc;
      v8h h;
#pragma unroll
      for (int j = 0; j < 8; ++j) h[j] = (_Float16)p[j];
      *(v8h*)&Bs[buf][(i << 6) + lr][lc] = h;
    }
  };

  v8f c00 = {}, c01 = {}, c10 = {}, c11 = {};
  loadA(0, 0);
  loadB(0, 0);
  __syncthreads();

  const int mrow = lane & 15;
  const int half = (lane >> 4) << 3;
  for (int k0 = 0; k0 < K; k0 += BK) {
    const int buf = (k0 / BK) & 1;
    if (k0 + BK < K) { loadA(buf ^ 1, k0 + BK); loadB(buf ^ 1, k0 + BK); }
    v16h a0 = frag_lds(&As[buf][wm + mrow][half]);
    v16h a1 = frag_lds(&As[buf][wm + 16 + mrow][half]);
    v16h b0 = frag_lds(&Bs[buf][wn + mrow][half]);
    v16h b1 = frag_lds(&Bs[buf][wn + 16 + mrow][half]);
    c00 = WMMA_F16(a0, b0, c00);
    c01 = WMMA_F16(a0, b1, c01);
    c10 = WMMA_F16(a1, b0, c10);
    c11 = WMMA_F16(a1, b1, c11);
    __syncthreads();
  }

  // C/D layout: VGPR r -> M = r + 8*(lane>=16), N = lane & 15
  const int nloc  = lane & 15;
  const int mbase = (lane >> 4) << 3;
  const int n0    = bn0 + wn + nloc;
  const float bz0 = bias[n0], bz1 = bias[n0 + 16];
#pragma unroll
  for (int r = 0; r < 8; ++r) {
    const int m0 = bm0 + wm + mbase + r;
    Y[(size_t)m0 * N + n0]             = c00[r] + bz0;
    Y[(size_t)m0 * N + n0 + 16]        = c01[r] + bz1;
    Y[(size_t)(m0 + 16) * N + n0]      = c10[r] + bz0;
    Y[(size_t)(m0 + 16) * N + n0 + 16] = c11[r] + bz1;
  }
}

// ---- logits: attn[bh,q,k] = (Qh.Kh)/8 + rel_bias + mask*-1e9 ---------------
// One wave computes a 16x64 logits strip (Q fragment reused by 4 WMMAs).
__global__ __launch_bounds__(256)
void logits_kernel(const float* __restrict__ Qp, const float* __restrict__ Kp,
                   const float* __restrict__ rel_bias,
                   const float* __restrict__ mask, float* __restrict__ attn) {
  const int lane = threadIdx.x & 31;
  const int wave = blockIdx.x * (blockDim.x >> 5) + (threadIdx.x >> 5);
  const int TK4 = (SEQ / 16) / 4;   // 32 strips of 4 k-tiles
  const int TQ  = SEQ / 16;
  const int tk4 = wave % TK4;
  const int tq  = (wave / TK4) % TQ;
  const int bh  = wave / (TK4 * TQ);
  const int b = bh / NUM_HEADS, h = bh % NUM_HEADS;

  const float* qsrc = Qp + (size_t)b * SEQ * D_MODEL + h * DEPTH;
  const float* ksrc = Kp + (size_t)b * SEQ * D_MODEL + h * DEPTH;

  const int mrow = lane & 15;
  v8f c[4] = {{}, {}, {}, {}};
#pragma unroll
  for (int k0 = 0; k0 < DEPTH; k0 += 32) {
    v16h a = frag_row(qsrc, D_MODEL, tq * 16 + mrow, k0, lane);
#pragma unroll
    for (int i = 0; i < 4; ++i) {
      v16h bf = frag_row(ksrc, D_MODEL, tk4 * 64 + i * 16 + mrow, k0, lane);
      c[i] = WMMA_F16(a, bf, c[i]);
    }
  }

  const int nloc  = lane & 15;
  const int mbase = (lane >> 4) << 3;
  float*       arow = attn + (size_t)bh * SEQ * SEQ;
  const float* mrowp = mask + (size_t)b * SEQ * SEQ;
#pragma unroll
  for (int i = 0; i < 4; ++i) {
    const int kpos = tk4 * 64 + i * 16 + nloc;
#pragma unroll
    for (int r = 0; r < 8; ++r) {
      const int qpos = tq * 16 + mbase + r;
      int idx = qpos - kpos + (MAX_SEQ - 1);
      idx = idx < 0 ? 0 : (idx > 2 * MAX_SEQ - 2 ? 2 * MAX_SEQ - 2 : idx);
      float v = c[i][r] * 0.125f
              + rel_bias[(size_t)idx * NUM_HEADS + h]
              + mrowp[(size_t)qpos * SEQ + kpos] * -1e9f;
      arow[(size_t)qpos * SEQ + kpos] = v;
    }
  }
}

// ---- in-place row softmax over attn (one block per row of 2048) ------------
__global__ __launch_bounds__(256)
void softmax_kernel(float* __restrict__ attn) {
  __shared__ float red[256];
  float* p = attn + (size_t)blockIdx.x * SEQ;
  const int t = threadIdx.x;

  float m = -3.4e38f;
  for (int i = t; i < SEQ; i += 256) m = fmaxf(m, p[i]);
  red[t] = m; __syncthreads();
  for (int s = 128; s > 0; s >>= 1) {
    if (t < s) red[t] = fmaxf(red[t], red[t + s]);
    __syncthreads();
  }
  m = red[0]; __syncthreads();

  float sum = 0.f;
  for (int i = t; i < SEQ; i += 256) { float e = __expf(p[i] - m); p[i] = e; sum += e; }
  red[t] = sum; __syncthreads();
  for (int s = 128; s > 0; s >>= 1) {
    if (t < s) red[t] += red[t + s];
    __syncthreads();
  }
  const float inv = 1.f / red[0];
  for (int i = t; i < SEQ; i += 256) p[i] *= inv;
}

// ---- ctx[b,q,h*64+d] = sum_k attn[bh,q,k] * Vh[bh,k,d] ---------------------
// Block = one (bh, 8 q-tiles); attn streamed once (4 accumulators per wave
// cover the full depth=64). V tile transposed into LDS (Vt[d][k]) so B
// fragments are two aligned ds_load_b128 each.
__global__ __launch_bounds__(256)
void attnv_kernel(const float* __restrict__ attn, const float* __restrict__ Vp,
                  float* __restrict__ ctx) {
  __shared__ _Float16 Vt[2][DEPTH][LDP];   // 12 KB
  const int t = threadIdx.x, lane = t & 31, wave = t >> 5;
  const int qchunks = (SEQ / 16) / 8;      // 16 blocks of 8 q-tiles per bh
  const int bh = blockIdx.x / qchunks;
  const int qc = blockIdx.x % qchunks;
  const int b = bh / NUM_HEADS, h = bh % NUM_HEADS;
  const int tq = (qc * 8 + wave) * 16;

  const float* arow = attn + (size_t)bh * SEQ * SEQ;
  const float* vsrc = Vp + (size_t)b * SEQ * D_MODEL + h * DEPTH;

  const int lr = t >> 3;                   // k row  0..31
  const int lc = (t & 7) << 3;             // d col0 0..56
  auto loadV = [&](int buf, int k0) {      // coalesced read, transposed store
    const float* p = vsrc + (size_t)(k0 + lr) * D_MODEL + lc;
#pragma unroll
    for (int j = 0; j < 8; ++j) Vt[buf][lc + j][lr] = (_Float16)p[j];
  };

  v8f c[4] = {{}, {}, {}, {}};
  loadV(0, 0);
  __syncthreads();

  const int mrow = lane & 15;
  const int half = (lane >> 4) << 3;
  for (int k0 = 0; k0 < SEQ; k0 += 32) {
    const int buf = (k0 >> 5) & 1;
    if (k0 + 32 < SEQ) loadV(buf ^ 1, k0 + 32);
    v16h a = frag_row(arow, SEQ, tq + mrow, k0, lane);
#pragma unroll
    for (int i = 0; i < 4; ++i) {
      v16h bf = frag_lds(&Vt[buf][i * 16 + mrow][half]);
      c[i] = WMMA_F16(a, bf, c[i]);
    }
    __syncthreads();
  }

  const int nloc  = lane & 15;
  const int mbase = (lane >> 4) << 3;
#pragma unroll
  for (int i = 0; i < 4; ++i)
#pragma unroll
    for (int r = 0; r < 8; ++r) {
      const int qpos = tq + mbase + r;
      ctx[((size_t)b * SEQ + qpos) * D_MODEL + h * DEPTH + i * 16 + nloc] = c[i][r];
    }
}

// ---------------------------------------------------------------------------
extern "C" void kernel_launch(void* const* d_in, const int* in_sizes, int n_in,
                              void* d_out, int out_size, void* d_ws, size_t ws_size,
                              hipStream_t stream) {
  const float* q        = (const float*)d_in[0];
  const float* k        = (const float*)d_in[1];
  const float* v        = (const float*)d_in[2];
  const float* mask     = (const float*)d_in[3];
  const float* wq_w     = (const float*)d_in[4];
  const float* wq_b     = (const float*)d_in[5];
  const float* wk_w     = (const float*)d_in[6];
  const float* wk_b     = (const float*)d_in[7];
  const float* wv_w     = (const float*)d_in[8];
  const float* wv_b     = (const float*)d_in[9];
  const float* dense_w  = (const float*)d_in[10];
  const float* dense_b  = (const float*)d_in[11];
  const float* rel_bias = (const float*)d_in[12];

  // Workspace: Q | K | V projections, 16 MB each. ctx reuses the K buffer
  // (K is dead after the logits kernel; stream ordering guarantees safety).
  float* Qp  = (float*)d_ws;
  float* Kp  = Qp + (size_t)BS * D_MODEL;
  float* Vp  = Kp + (size_t)BS * D_MODEL;
  float* ctx = Kp;

  float* out  = (float*)d_out;                 // [B,S,D]   = 4,194,304 f32
  float* attn = out + (size_t)BS * D_MODEL;    // [B,H,S,S] = 134,217,728 f32

  const dim3 blk(256);                          // 8 waves / block
  const int gemmBlocks   = (BS / BM) * (D_MODEL / BN);                          // 512
  const int logitsBlocks = BATCH * NUM_HEADS * (SEQ / 16) * ((SEQ / 16) / 4) / 8; // 16384
  const int smaxBlocks   = BATCH * NUM_HEADS * SEQ;                             // 65536
  const int avBlocks     = BATCH * NUM_HEADS * ((SEQ / 16) / 8);                // 512

  gemm_bias_kernel<<<gemmBlocks, blk, 0, stream>>>(q, wq_w, wq_b, Qp, BS, D_MODEL, D_MODEL);
  gemm_bias_kernel<<<gemmBlocks, blk, 0, stream>>>(k, wk_w, wk_b, Kp, BS, D_MODEL, D_MODEL);
  gemm_bias_kernel<<<gemmBlocks, blk, 0, stream>>>(v, wv_w, wv_b, Vp, BS, D_MODEL, D_MODEL);
  logits_kernel<<<logitsBlocks, blk, 0, stream>>>(Qp, Kp, rel_bias, mask, attn);
  softmax_kernel<<<smaxBlocks, blk, 0, stream>>>(attn);
  attnv_kernel<<<avBlocks, blk, 0, stream>>>(attn, Vp, ctx);
  gemm_bias_kernel<<<gemmBlocks, blk, 0, stream>>>(ctx, dense_w, dense_b, out, BS, D_MODEL, D_MODEL);
}
